// ParaTransformerDecoderLayer_446676599077
// MI455X (gfx1250) — matrix operand order repaired and checked by reading, not compiled, not run on hardware
//
#include <hip/hip_runtime.h>
#include <hip/hip_bf16.h>
#include <stdint.h>

// ---------------- problem dims ----------------
#define B_DIM 16
#define T_SEQ 120
#define N_J   24
#define D_MOD 256
#define H_HEAD 8
#define DP_H  32
#define MRP_  32
#define TP    128            // T padded to WMMA tiles
#define RJ    65             // 2*MRP+1 rel entries
#define RJN   80             // rel padded as N-dim (5 tiles)
#define RJK   96             // rel padded as K-dim (3 k-steps)
#define BT_ROWS (B_DIM*T_SEQ)      // 1920
#define M_ALL   (B_DIM*T_SEQ*N_J)  // 46080
#define SCALE_ATTN 0.17677669529663687f  // 1/sqrt(32)

typedef __attribute__((ext_vector_type(16))) _Float16 v16h;
typedef __attribute__((ext_vector_type(8)))  float    v8f;
typedef __attribute__((ext_vector_type(4)))  int      v4i_;

#define WMMA_F16(c, a, b) \
  __builtin_amdgcn_wmma_f32_16x16x32_f16(false, (a), false, (b), (short)0, (c), false, false)

union HF16 { v16h v; uint4 q[2]; };

// ---- gfx1250 async global->LDS copy (ASYNCcnt path), with sync fallback ----
#if __has_builtin(__builtin_amdgcn_global_load_async_to_lds_b128)
#define HAS_ASYNC_LDS 1
typedef __attribute__((address_space(1))) v4i_* gp128_t;  // global int4*
typedef __attribute__((address_space(3))) v4i_* lp128_t;  // LDS int4*
#else
#define HAS_ASYNC_LDS 0
#endif

__device__ __forceinline__ void async_copy16(const _Float16* g, _Float16* l) {
#if HAS_ASYNC_LDS
  __builtin_amdgcn_global_load_async_to_lds_b128((gp128_t)(g), (lp128_t)(l), 0, 0);
#else
  *(uint4*)l = *(const uint4*)g;
#endif
}
__device__ __forceinline__ void async_join() {
#if HAS_ASYNC_LDS
  asm volatile("s_wait_asynccnt 0" ::: "memory");
#endif
}

// A-matrix 16x32 f16 fragment (row-major source):
// lane l holds row m=l%16; k = 16*(e/8) + 8*(l/16) + e%8  -> two 16B chunks.
__device__ __forceinline__ v16h frag_a_h(const _Float16* row, int kstart, int hi) {
  const _Float16* p = row + kstart + hi * 8;
  HF16 f;
  f.q[0] = *(const uint4*)(p);
  f.q[1] = *(const uint4*)(p + 16);
  return f.v;
}
// B-matrix 32x16 f16 fragment from B-transposed storage BT[n][k]:
// lane l holds col n=l%16; k = 16*(e/8) + 8*(l/16) + e%8 -> two 16B chunks.
__device__ __forceinline__ v16h frag_b_h(const _Float16* btrow, int kstart, int hi) {
  const _Float16* p = btrow + kstart + hi * 8;
  HF16 f;
  f.q[0] = *(const uint4*)(p);
  f.q[1] = *(const uint4*)(p + 16);
  return f.v;
}
// A fragment converted on the fly from an f32 row (for exp(S) and Ar)
__device__ __forceinline__ v16h frag_a_f(const float* row, int kstart, int hi) {
  const float* p = row + kstart + hi * 8;
  v16h h;
#pragma unroll
  for (int i = 0; i < 8; ++i) h[i] = (_Float16)p[i];
#pragma unroll
  for (int i = 0; i < 8; ++i) h[8 + i] = (_Float16)p[16 + i];
  return h;
}

// ---------------- elementwise converts ----------------
__global__ void f32_to_f16_kernel(const float* __restrict__ src, _Float16* __restrict__ dst,
                                  long long ntot) {
  long long i = (long long)blockIdx.x * blockDim.x + threadIdx.x;
  if (i < ntot) dst[i] = (_Float16)src[i];
}

// src[b][k][n] (f32) -> dst[b][n][k] (f16)   (weight transpose for BT layout)
__global__ void trans_f16_kernel(const float* __restrict__ src, _Float16* __restrict__ dst,
                                 int K, int N, long long ntot) {
  long long i = (long long)blockIdx.x * blockDim.x + threadIdx.x;
  if (i < ntot) {
    long long kn = (long long)K * N;
    long long b = i / kn;
    long long rem = i - b * kn;
    int k = (int)(rem / N);
    int nn = (int)(rem % N);
    dst[b * kn + (long long)nn * K + k] = (_Float16)src[i];
  }
}

// ---------------- batched strided WMMA GEMM (K must be 256) ----------------
// C[m][n] = sum_k A[m][k] * BT[n][k] + bias[n] (+resid) (opt relu), per batch z.
// Tile: 64x64 per 128-thread block; A panel (64x256) async-staged to LDS once,
// shared by all 4 waves; each wave owns an n-strip of 16 with 4 M-tiles.
__global__ void __launch_bounds__(128)
gemm_wmma_kernel(const _Float16* __restrict__ A, long long a_bs, long long a_rs,
                 const _Float16* __restrict__ BT, long long b_bs,
                 const float* __restrict__ bias, long long bias_bs,
                 _Float16* __restrict__ C16, long long c16_bs, long long c16_rs,
                 float* __restrict__ C32, long long c32_bs, long long c32_rs,
                 const float* __restrict__ R32, long long r_bs, long long r_rs,
                 int relu) {
  __shared__ _Float16 sA[64 * D_MOD];  // 32 KB

  const int w = threadIdx.x >> 5, lane = threadIdx.x & 31;
  const int hi = lane >> 4, ln = lane & 15;
  const int m0 = blockIdx.x * 64;
  const int n0 = blockIdx.y * 64 + w * 16;
  const long long bz = blockIdx.z;
  const int col = n0 + ln;

  const _Float16* Ab = A + bz * a_bs;
  const _Float16* Bb = BT + bz * b_bs + (long long)col * D_MOD;

  // async-stage the 64x256 A panel into LDS (16B chunks; 2048 chunks total)
#pragma unroll
  for (int it = 0; it < 16; ++it) {
    int idx = it * 128 + threadIdx.x;
    int r = idx >> 5, q = idx & 31;  // row in panel, 16B-chunk within row
    async_copy16(Ab + (long long)(m0 + r) * a_rs + q * 8, sA + r * D_MOD + q * 8);
  }
  async_join();
  __syncthreads();

  v8f zero = {0.f, 0.f, 0.f, 0.f, 0.f, 0.f, 0.f, 0.f};
  v8f acc[4];
#pragma unroll
  for (int i = 0; i < 4; ++i) acc[i] = zero;

  for (int kk = 0; kk < D_MOD; kk += 32) {
    v16h bf = frag_b_h(Bb, kk, hi);
#pragma unroll
    for (int mt = 0; mt < 4; ++mt) {
      v16h af = frag_a_h(sA + (mt * 16 + ln) * D_MOD, kk, hi);
      acc[mt] = WMMA_F16(acc[mt], af, bf);
    }
  }

  float bv = bias ? bias[bz * bias_bs + col] : 0.0f;
#pragma unroll
  for (int mt = 0; mt < 4; ++mt) {
#pragma unroll
    for (int r = 0; r < 8; ++r) {
      long long m = m0 + mt * 16 + r + 8 * hi;
      float v = acc[mt][r] + bv;
      if (R32) v += R32[bz * r_bs + m * r_rs + col];
      if (relu) v = fmaxf(v, 0.0f);
      if (C32) C32[bz * c32_bs + m * c32_rs + col] = v;
      if (C16) C16[bz * c16_bs + m * c16_rs + col] = (_Float16)v;
    }
  }
}

// ---------------- temporal attention (per b,n,h) ----------------
// LDS layout (bytes):
#define OFF_Q     0         // 128x32 f16 = 8192
#define OFF_K     8192      // 8192
#define OFF_VT    16384     // V^T: 32x128 f16 = 8192
#define OFF_RK    24576     // relK: 80x32 f16 = 5120
#define OFF_RVT   29696     // relV^T: 32x96 f16 = 6144
#define OFF_S     35840     // S/exp: 128x128 f32 = 65536
#define OFF_R     101376    // R: 128x80 f32 = 40960
#define OFF_AR    142336    // Ar: 128x96 f32 = 49152
#define OFF_INV   191488    // 128 f32 = 512
#define SMEM_BYTES 192000

__global__ void __launch_bounds__(128)
temporal_attn_kernel(const _Float16* __restrict__ Qg, const _Float16* __restrict__ Kg,
                     const _Float16* __restrict__ Vg, const _Float16* __restrict__ relK,
                     const _Float16* __restrict__ relV, const float* __restrict__ mask,
                     _Float16* __restrict__ Og) {
  extern __shared__ char smem[];
  _Float16* sQ  = (_Float16*)(smem + OFF_Q);
  _Float16* sK  = (_Float16*)(smem + OFF_K);
  _Float16* sVT = (_Float16*)(smem + OFF_VT);
  _Float16* sRK = (_Float16*)(smem + OFF_RK);
  _Float16* sRVT= (_Float16*)(smem + OFF_RVT);
  float* sS   = (float*)(smem + OFF_S);
  float* sR   = (float*)(smem + OFF_R);
  float* sAr  = (float*)(smem + OFF_AR);
  float* sInv = (float*)(smem + OFF_INV);

  const int tid = threadIdx.x;
  const int w = tid >> 5, lane = tid & 31, hi = lane >> 4, ln = lane & 15;
  const int h = blockIdx.x, n = blockIdx.y, b = blockIdx.z;

  const size_t rowbase = (size_t)(n * BT_ROWS + b * T_SEQ);
  const _Float16* qg = Qg + rowbase * D_MOD + h * DP_H;
  const _Float16* kg = Kg + rowbase * D_MOD + h * DP_H;
  const _Float16* vg = Vg + rowbase * D_MOD + h * DP_H;

  // ---- Phase A: stage Q, K (async), V^T, rel tables into LDS (zero pad) ----
  {
    int r = tid;  // 0..127
    if (r < T_SEQ) {
      // Q,K rows: 64B each -> 4 async 16B chunks per row
#pragma unroll
      for (int i = 0; i < 4; ++i) {
        async_copy16(qg + (size_t)r * D_MOD + i * 8, sQ + r * DP_H + i * 8);
        async_copy16(kg + (size_t)r * D_MOD + i * 8, sK + r * DP_H + i * 8);
      }
      const _Float16* vr = vg + (size_t)r * D_MOD;
#pragma unroll
      for (int c = 0; c < DP_H; ++c) sVT[c * TP + r] = vr[c];
    } else {
      uint4 z; z.x = z.y = z.z = z.w = 0u;
      uint4* dq = (uint4*)(sQ + r * DP_H);
      uint4* dk = (uint4*)(sK + r * DP_H);
#pragma unroll
      for (int i = 0; i < 4; ++i) { dq[i] = z; dk[i] = z; }
#pragma unroll
      for (int c = 0; c < DP_H; ++c) sVT[c * TP + r] = (_Float16)0.0f;
    }
    if (tid < RJN) {
      int j = tid;
#pragma unroll
      for (int c = 0; c < DP_H; ++c)
        sRK[j * DP_H + c] = (j < RJ) ? relK[j * DP_H + c] : (_Float16)0.0f;
    }
    if (tid < RJK) {
      int j = tid;
#pragma unroll
      for (int c = 0; c < DP_H; ++c)
        sRVT[c * RJK + j] = (j < RJ) ? relV[j * DP_H + c] : (_Float16)0.0f;
    }
  }
  async_join();
  __syncthreads();

  // ---- Phase B: S = Q K^T (8x8 tiles), R = Q relK^T (8x5 tiles), WMMA ----
  for (int mt = w; mt < 8; mt += 4) {
    v16h af = frag_a_h(sQ + (mt * 16 + ln) * DP_H, 0, hi);
    for (int nt = 0; nt < 8; ++nt) {
      v16h bf = frag_b_h(sK + (nt * 16 + ln) * DP_H, 0, hi);
      v8f acc = {0.f, 0.f, 0.f, 0.f, 0.f, 0.f, 0.f, 0.f};
      acc = WMMA_F16(acc, af, bf);
#pragma unroll
      for (int r = 0; r < 8; ++r)
        sS[(mt * 16 + r + 8 * hi) * TP + nt * 16 + ln] = acc[r];
    }
    for (int jt = 0; jt < 5; ++jt) {
      v16h bf = frag_b_h(sRK + (jt * 16 + ln) * DP_H, 0, hi);
      v8f acc = {0.f, 0.f, 0.f, 0.f, 0.f, 0.f, 0.f, 0.f};
      acc = WMMA_F16(acc, af, bf);
#pragma unroll
      for (int r = 0; r < 8; ++r)
        sR[(mt * 16 + r + 8 * hi) * RJN + jt * 16 + ln] = acc[r];
    }
  }
  __syncthreads();

  // ---- Phase C: per-row softmax (keep unnormalized exp in sS), build Ar ----
  {
    int t = tid;
    float* srow = sS + t * TP;
    float* rrow = sR + t * RJN;
    float* arow = sAr + t * RJK;
    for (int j = 0; j < RJK; ++j) arow[j] = 0.0f;
    if (t < T_SEQ) {
      const float* mrow = mask + t * T_SEQ;
      float mx = -3.0e38f;
      for (int s = 0; s < T_SEQ; ++s) {
        int j = s - t;
        j = (j < -MRP_) ? -MRP_ : ((j > MRP_) ? MRP_ : j);
        j += MRP_;
        float l = (srow[s] + rrow[j]) * SCALE_ATTN + mrow[s] * (-1.0e9f);
        srow[s] = l;
        mx = fmaxf(mx, l);
      }
      float sum = 0.0f;
      for (int s = 0; s < T_SEQ; ++s) {
        float e = __expf(srow[s] - mx);
        srow[s] = e;
        sum += e;
        int j = s - t;
        j = (j < -MRP_) ? -MRP_ : ((j > MRP_) ? MRP_ : j);
        arow[j + MRP_] += e;
      }
      for (int s = T_SEQ; s < TP; ++s) srow[s] = 0.0f;
      sInv[t] = 1.0f / sum;
    } else {
      for (int s = 0; s < TP; ++s) srow[s] = 0.0f;
      sInv[t] = 1.0f;
    }
  }
  __syncthreads();

  // ---- Phase D: O = exp(S)*V + Ar*relV (WMMA chains), normalize, store ----
  for (int mt = w; mt < 8; mt += 4) {
    const float* arowA = sS + (mt * 16 + ln) * TP;
    const float* arowR = sAr + (mt * 16 + ln) * RJK;
    for (int ct = 0; ct < 2; ++ct) {
      v8f acc = {0.f, 0.f, 0.f, 0.f, 0.f, 0.f, 0.f, 0.f};
#pragma unroll
      for (int ks = 0; ks < 4; ++ks) {
        v16h af = frag_a_f(arowA, ks * 32, hi);
        v16h bf = frag_b_h(sVT + (ct * 16 + ln) * TP, ks * 32, hi);
        acc = WMMA_F16(acc, af, bf);
      }
#pragma unroll
      for (int ks = 0; ks < 3; ++ks) {
        v16h af = frag_a_f(arowR, ks * 32, hi);
        v16h bf = frag_b_h(sRVT + (ct * 16 + ln) * RJK, ks * 32, hi);
        acc = WMMA_F16(acc, af, bf);
      }
#pragma unroll
      for (int r = 0; r < 8; ++r) {
        int t = mt * 16 + r + 8 * hi;
        if (t < T_SEQ) {
          float v = acc[r] * sInv[t];
          Og[((size_t)((b * T_SEQ + t) * N_J + n)) * D_MOD + h * DP_H + ct * 16 + ln] =
              (_Float16)v;
        }
      }
    }
  }
}

// ---------------- spatial attention (per b,t): 24 joints, VALU from LDS ----------------
__global__ void __launch_bounds__(256)
spatial_attn_kernel(const _Float16* __restrict__ Qg, const _Float16* __restrict__ Kg,
                    const _Float16* __restrict__ Vg, _Float16* __restrict__ Og) {
  __shared__ _Float16 sQ[N_J * D_MOD];
  __shared__ _Float16 sK[N_J * D_MOD];
  __shared__ _Float16 sV[N_J * D_MOD];
  const int tid = threadIdx.x;
  const size_t base = (size_t)blockIdx.x * (N_J * D_MOD);
  // async-stage q/k/v for this (b,t): 3 x 12KB in 16B chunks (768 chunks each)
  for (int i = tid; i < N_J * D_MOD / 8; i += 256) {
    async_copy16(Qg + base + i * 8, sQ + i * 8);
    async_copy16(Kg + base + i * 8, sK + i * 8);
    async_copy16(Vg + base + i * 8, sV + i * 8);
  }
  async_join();
  __syncthreads();
  if (tid < N_J * H_HEAD) {
    const int nj = tid % N_J, hh = tid / N_J;
    const int bc = hh * DP_H;
    float q[DP_H];
#pragma unroll
    for (int c = 0; c < DP_H; ++c) q[c] = (float)sQ[nj * D_MOD + bc + c];
    float lg[N_J];
    float mx = -3.0e38f;
    for (int m = 0; m < N_J; ++m) {
      float d = 0.0f;
#pragma unroll
      for (int c = 0; c < DP_H; ++c) d += q[c] * (float)sK[m * D_MOD + bc + c];
      d *= SCALE_ATTN;
      lg[m] = d;
      mx = fmaxf(mx, d);
    }
    float sum = 0.0f;
    for (int m = 0; m < N_J; ++m) { float e = __expf(lg[m] - mx); lg[m] = e; sum += e; }
    float inv = 1.0f / sum;
    float o[DP_H];
#pragma unroll
    for (int c = 0; c < DP_H; ++c) o[c] = 0.0f;
    for (int m = 0; m < N_J; ++m) {
      float pm = lg[m] * inv;
#pragma unroll
      for (int c = 0; c < DP_H; ++c) o[c] += pm * (float)sV[m * D_MOD + bc + c];
    }
    _Float16* orow = Og + base + nj * D_MOD + bc;
#pragma unroll
    for (int c = 0; c < DP_H; ++c) orow[c] = (_Float16)o[c];
  }
}

// ---------------- host orchestration ----------------
extern "C" void kernel_launch(void* const* d_in, const int* in_sizes, int n_in,
                              void* d_out, int out_size, void* d_ws, size_t ws_size,
                              hipStream_t stream) {
  (void)in_sizes; (void)n_in; (void)out_size; (void)ws_size;
  const float* x      = (const float*)d_in[0];
  const float* mask   = (const float*)d_in[1];
  const float* wq_t   = (const float*)d_in[2];
  const float* wk_t   = (const float*)d_in[3];
  const float* wv_t   = (const float*)d_in[4];
  const float* bq_t   = (const float*)d_in[5];
  const float* bk_t   = (const float*)d_in[6];
  const float* bv_t   = (const float*)d_in[7];
  const float* wo_t   = (const float*)d_in[8];
  const float* bo_t   = (const float*)d_in[9];
  const float* rel_key= (const float*)d_in[10];
  const float* rel_val= (const float*)d_in[11];
  const float* wq_s   = (const float*)d_in[12];
  const float* wk_s   = (const float*)d_in[13];
  const float* wv_s   = (const float*)d_in[14];
  const float* wo_s   = (const float*)d_in[15];
  const float* bq_s   = (const float*)d_in[16];
  const float* bk_s   = (const float*)d_in[17];
  const float* bv_s   = (const float*)d_in[18];
  const float* bo_s   = (const float*)d_in[19];
  const float* ff1_w  = (const float*)d_in[20];
  const float* ff1_b  = (const float*)d_in[21];
  const float* ff2_w  = (const float*)d_in[22];
  const float* ff2_b  = (const float*)d_in[23];
  float* outp = (float*)d_out;

  // workspace allocator
  char* wsp = (char*)d_ws;
  auto alloc = [&](size_t bytes) -> char* {
    char* r = wsp;
    wsp += (bytes + 255) & ~(size_t)255;
    return r;
  };
  const size_t EL   = (size_t)M_ALL * D_MOD;          // 46080*256
  const size_t WJNT = (size_t)N_J * D_MOD * D_MOD;    // 24*65536

  _Float16* xh   = (_Float16*)alloc(EL * 2);
  _Float16* WqtT = (_Float16*)alloc(WJNT * 2);
  _Float16* WktT = (_Float16*)alloc(WJNT * 2);
  _Float16* WvtT = (_Float16*)alloc(WJNT * 2);
  _Float16* WotT = (_Float16*)alloc((size_t)D_MOD * D_MOD * 2);
  _Float16* WqsT = (_Float16*)alloc((size_t)D_MOD * D_MOD * 2);
  _Float16* WksT = (_Float16*)alloc((size_t)D_MOD * D_MOD * 2);
  _Float16* WvsT = (_Float16*)alloc((size_t)D_MOD * D_MOD * 2);
  _Float16* WosT = (_Float16*)alloc((size_t)D_MOD * D_MOD * 2);
  _Float16* Ff1T = (_Float16*)alloc(WJNT * 2);
  _Float16* Ff2T = (_Float16*)alloc(WJNT * 2);
  _Float16* relKh = (_Float16*)alloc((size_t)RJ * DP_H * 2);
  _Float16* relVh = (_Float16*)alloc((size_t)RJ * DP_H * 2);
  _Float16* Qb   = (_Float16*)alloc(EL * 2);   // temporal Q, reused as spatial q
  _Float16* Kb   = (_Float16*)alloc(EL * 2);
  _Float16* Vb   = (_Float16*)alloc(EL * 2);
  _Float16* attT = (_Float16*)alloc(EL * 2);   // temporal attn out, reused as FFN hidden
  _Float16* satt = (_Float16*)alloc(EL * 2);
  _Float16* yh   = (_Float16*)alloc(EL * 2);
  float*    yf   = (float*)alloc(EL * 4);

  // ---- converts ----
  {
    long long n = (long long)EL;
    f32_to_f16_kernel<<<(unsigned)((n + 255) / 256), 256, 0, stream>>>(x, xh, n);
  }
  {
    long long n = (long long)RJ * DP_H;
    f32_to_f16_kernel<<<(unsigned)((n + 255) / 256), 256, 0, stream>>>(rel_key, relKh, n);
    f32_to_f16_kernel<<<(unsigned)((n + 255) / 256), 256, 0, stream>>>(rel_val, relVh, n);
  }
  auto wtrans = [&](const float* src, _Float16* dst, int batch) {
    long long n = (long long)batch * D_MOD * D_MOD;
    trans_f16_kernel<<<(unsigned)((n + 255) / 256), 256, 0, stream>>>(src, dst, D_MOD, D_MOD, n);
  };
  wtrans(wq_t, WqtT, N_J);  wtrans(wk_t, WktT, N_J);  wtrans(wv_t, WvtT, N_J);
  wtrans(wo_t, WotT, 1);
  wtrans(wq_s, WqsT, 1);    wtrans(wk_s, WksT, 1);    wtrans(wv_s, WvsT, 1);
  wtrans(wo_s, WosT, 1);
  wtrans(ff1_w, Ff1T, N_J); wtrans(ff2_w, Ff2T, N_J);

  auto gemm = [&](int mtiles, int batch,
                  const _Float16* A, long long a_bs, long long a_rs,
                  const _Float16* BT, long long b_bs,
                  const float* bias, long long bias_bs,
                  _Float16* C16, long long c16_bs, long long c16_rs,
                  float* C32, long long c32_bs, long long c32_rs,
                  const float* R32, long long r_bs, long long r_rs, int relu) {
    gemm_wmma_kernel<<<dim3(mtiles, D_MOD / 64, batch), 128, 0, stream>>>(
        A, a_bs, a_rs, BT, b_bs, bias, bias_bs, C16, c16_bs, c16_rs,
        C32, c32_bs, c32_rs, R32, r_bs, r_rs, relu);
  };
  const long long JBATCH = (long long)BT_ROWS * D_MOD;  // 491520

  // ---- temporal QKV (per joint) ----
  gemm(BT_ROWS / 64, N_J, xh, D_MOD, (long long)N_J * D_MOD, WqtT, D_MOD * D_MOD,
       bq_t, D_MOD, Qb, JBATCH, D_MOD, nullptr, 0, 0, nullptr, 0, 0, 0);
  gemm(BT_ROWS / 64, N_J, xh, D_MOD, (long long)N_J * D_MOD, WktT, D_MOD * D_MOD,
       bk_t, D_MOD, Kb, JBATCH, D_MOD, nullptr, 0, 0, nullptr, 0, 0, 0);
  gemm(BT_ROWS / 64, N_J, xh, D_MOD, (long long)N_J * D_MOD, WvtT, D_MOD * D_MOD,
       bv_t, D_MOD, Vb, JBATCH, D_MOD, nullptr, 0, 0, nullptr, 0, 0, 0);

  // ---- temporal attention ----
  (void)hipFuncSetAttribute((const void*)temporal_attn_kernel,
                            hipFuncAttributeMaxDynamicSharedMemorySize, SMEM_BYTES);
  temporal_attn_kernel<<<dim3(H_HEAD, N_J, B_DIM), 128, SMEM_BYTES, stream>>>(
      Qb, Kb, Vb, relKh, relVh, mask, attT);

  // ---- y = x + attT*wo_t + bo_t ----
  gemm(M_ALL / 64, 1, attT, 0, D_MOD, WotT, 0, bo_t, 0,
       nullptr, 0, 0, yf, 0, D_MOD, x, 0, D_MOD, 0);

  // ---- spatial QKV (shared linears) ----
  gemm(M_ALL / 64, 1, xh, 0, D_MOD, WqsT, 0, bq_s, 0, Qb, 0, D_MOD,
       nullptr, 0, 0, nullptr, 0, 0, 0);
  gemm(M_ALL / 64, 1, xh, 0, D_MOD, WksT, 0, bk_s, 0, Kb, 0, D_MOD,
       nullptr, 0, 0, nullptr, 0, 0, 0);
  gemm(M_ALL / 64, 1, xh, 0, D_MOD, WvsT, 0, bv_s, 0, Vb, 0, D_MOD,
       nullptr, 0, 0, nullptr, 0, 0, 0);

  // ---- spatial attention ----
  spatial_attn_kernel<<<BT_ROWS, 256, 0, stream>>>(Qb, Kb, Vb, satt);

  // ---- y += satt*wo_s + bo_s ; emit yh (f16) ----
  gemm(M_ALL / 64, 1, satt, 0, D_MOD, WosT, 0, bo_s, 0, yh, 0, D_MOD,
       yf, 0, D_MOD, yf, 0, D_MOD, 0);

  // ---- FFN1 (per joint, relu) -> hidden in attT buffer ----
  gemm(BT_ROWS / 64, N_J, yh, D_MOD, (long long)N_J * D_MOD, Ff1T, D_MOD * D_MOD,
       ff1_b, D_MOD, attT, JBATCH, D_MOD, nullptr, 0, 0, nullptr, 0, 0, 1);

  // ---- FFN2 (per joint) + residual y -> d_out [b,t,n,d] f32 ----
  gemm(BT_ROWS / 64, N_J, attT, JBATCH, D_MOD, Ff2T, D_MOD * D_MOD,
       ff2_b, D_MOD, nullptr, 0, 0,
       outp, D_MOD, (long long)N_J * D_MOD,
       yf, D_MOD, (long long)N_J * D_MOD, 0);
}
